// CrossResidueBlock_28054726378130
// MI455X (gfx1250) — matrix operand order repaired and checked by reading, not compiled, not run on hardware
//
#include <hip/hip_runtime.h>

typedef __attribute__((ext_vector_type(16))) _Float16 v16h;
typedef __attribute__((ext_vector_type(8)))  float    v8f;

#define D_   256
#define H_   8
#define DH_  32
#define A_   14
#define FF_  1024
#define NEG_BIG (-3.0e38f)

// ---------------------------------------------------------------------------
// WMMA fragment loaders, this code's own k order (16-bit A 16x32, B 32x16).
// A: lane m=L%16; element e -> K = e + 16*(L>=16)  (contiguous per lane).
// B: lane n=L%16; element e -> K = e + 16*(L>=16)  (contiguous per lane).
// ---------------------------------------------------------------------------
__device__ __forceinline__ v16h load_a_frag(const _Float16* __restrict__ base,
                                            int ld, int lane) {
  int m = lane & 15;
  int kofs = (lane & 16) ? 16 : 0;
  const _Float16* row = base + (size_t)m * ld + kofs;
  v16h a;
#pragma unroll
  for (int p = 0; p < 8; ++p) {
    int k = 2 * p;
    a[2 * p]     = row[k];
    a[2 * p + 1] = row[k + 1];
  }
  return a;
}

__device__ __forceinline__ v16h load_b_frag(const _Float16* __restrict__ base,
                                            int ld, int lane) {
  int n = lane & 15;
  int k0 = (lane & 16) ? 16 : 0;
  const _Float16* row = base + (size_t)n * ld + k0;
  v16h b;
#pragma unroll
  for (int e = 0; e < 16; ++e) b[e] = row[e];
  return b;
}

// ---------------------------------------------------------------------------
// Generic flat GEMM: out[M,N] = act[M,K] @ W[N,K]^T (torch Linear layout).
// EPI: 0 = store f16; 1 = +bias, exact GELU, store f16;
//      2 = +resid(fp32), store fp32; 3 = +bias +resid, store fp32.
// Wave computes a 32x64 tile (2 A-frags share each B-frag -> 8 WMMA/k-step);
// block = 8 waves = 256 rows x 64 cols.
// ---------------------------------------------------------------------------
template <int EPI>
__global__ void __launch_bounds__(256) gemm_wmma_kernel(
    const _Float16* __restrict__ act, const _Float16* __restrict__ W,
    const float* __restrict__ bias, const float* __restrict__ resid,
    _Float16* __restrict__ outh, float* __restrict__ outf,
    int M, int N, int K) {
  int lane = threadIdx.x & 31;
  int wave = threadIdx.x >> 5;
  int m0   = (blockIdx.x * 8 + wave) * 32;
  int n0   = blockIdx.y * 64;
  const _Float16* abase0 = act + (size_t)m0 * K;
  const _Float16* abase1 = act + (size_t)(m0 + 16) * K;

  v8f acc[2][4];
#pragma unroll
  for (int s = 0; s < 2; ++s)
#pragma unroll
    for (int j = 0; j < 4; ++j)
#pragma unroll
      for (int i = 0; i < 8; ++i) acc[s][j][i] = 0.0f;

  for (int kk = 0; kk < K; kk += 32) {
    v16h a0 = load_a_frag(abase0 + kk, K, lane);
    v16h a1 = load_a_frag(abase1 + kk, K, lane);
#pragma unroll
    for (int j = 0; j < 4; ++j) {
      const _Float16* wb = W + (size_t)(n0 + j * 16) * K + kk;
      if (kk + 32 < K)  // prefetch next k-step of this weight row group
        __builtin_prefetch(wb + 32, 0, 1);
      v16h b = load_b_frag(wb, K, lane);
      acc[0][j] = __builtin_amdgcn_wmma_f32_16x16x32_f16(false, a0, false, b,
                                                         (short)0, acc[0][j], false, false);
      acc[1][j] = __builtin_amdgcn_wmma_f32_16x16x32_f16(false, a1, false, b,
                                                         (short)0, acc[1][j], false, false);
    }
  }

  int ncol = lane & 15;
  int hi8  = (lane & 16) ? 8 : 0;
#pragma unroll
  for (int s = 0; s < 2; ++s) {
    int mrow0 = m0 + s * 16 + hi8;
#pragma unroll
    for (int j = 0; j < 4; ++j) {
      int n = n0 + j * 16 + ncol;
      float bv = (EPI == 1 || EPI == 3) ? bias[n] : 0.0f;
#pragma unroll
      for (int i = 0; i < 8; ++i) {
        int m = mrow0 + i;
        float v = acc[s][j][i] + bv;
        if (EPI == 1) v = 0.5f * v * (1.0f + erff(v * 0.70710678118654752f));
        if (EPI == 2 || EPI == 3) {
          v += resid[(size_t)m * N + n];
          outf[(size_t)m * N + n] = v;
        } else {
          outh[(size_t)m * N + n] = (_Float16)v;
        }
      }
    }
  }
}

// ---------------------------------------------------------------------------
// LayerNorm over D=256 + cast to f16. One wave per row (wave32).
// ---------------------------------------------------------------------------
__global__ void __launch_bounds__(256) ln_cast_kernel(
    const float* __restrict__ x, const float* __restrict__ w,
    const float* __restrict__ b, _Float16* __restrict__ out, int rows) {
  int lane = threadIdx.x & 31;
  int row  = (int)((blockIdx.x * blockDim.x + threadIdx.x) >> 5);
  if (row >= rows) return;
  const float* r = x + (size_t)row * D_;
  float v[8]; float s = 0.f;
#pragma unroll
  for (int i = 0; i < 8; ++i) { v[i] = r[lane + i * 32]; s += v[i]; }
#pragma unroll
  for (int m = 1; m < 32; m <<= 1) s += __shfl_xor(s, m, 32);
  float mean = s * (1.0f / D_);
  float qv = 0.f;
#pragma unroll
  for (int i = 0; i < 8; ++i) { float d = v[i] - mean; qv += d * d; }
#pragma unroll
  for (int m = 1; m < 32; m <<= 1) qv += __shfl_xor(qv, m, 32);
  float inv = rsqrtf(qv * (1.0f / D_) + 1e-5f);
  _Float16* o = out + (size_t)row * D_;
#pragma unroll
  for (int i = 0; i < 8; ++i) {
    int c = lane + i * 32;
    o[c] = (_Float16)((v[i] - mean) * inv * w[c] + b[c]);
  }
}

// ---------------------------------------------------------------------------
// koff/voff[s][e] = offset_embed[s] . {k_w,v_w}[e][:]  -> stored as f16
// ---------------------------------------------------------------------------
__global__ void offproj_kernel(const float* __restrict__ off,
                               const float* __restrict__ kw,
                               const float* __restrict__ vw,
                               _Float16* __restrict__ koff, _Float16* __restrict__ voff) {
  int s = blockIdx.x, e = threadIdx.x;
  const float* o = off + s * D_;
  float sk = 0.f, sv = 0.f;
  for (int d = 0; d < D_; ++d) {
    float ov = o[d];
    sk += ov * kw[(size_t)e * D_ + d];
    sv += ov * vw[(size_t)e * D_ + d];
  }
  koff[s * D_ + e] = (_Float16)sk;
  voff[s * D_ + e] = (_Float16)sv;
}

__global__ void cvt_f16_kernel(const float* __restrict__ in,
                               _Float16* __restrict__ out, int n) {
  int i = blockIdx.x * blockDim.x + threadIdx.x;
  if (i < n) out[i] = (_Float16)in[i];
}

// ---------------------------------------------------------------------------
// Sliding-window attention. One block per residue, one wave per head.
// scores(16x48) = Q(16x32) @ K^T ; softmax in C-layout via half-wave shuffles;
// O(16x32) = P(16x64 zero-padded) @ V(64x32), P & V^T staged in LDS.
// Slot offsets added as packed f16.
// ---------------------------------------------------------------------------
__global__ void __launch_bounds__(256) attn_kernel(
    const _Float16* __restrict__ qg, const _Float16* __restrict__ kg,
    const _Float16* __restrict__ vg, const _Float16* __restrict__ koff,
    const _Float16* __restrict__ voff, const int* __restrict__ amask,
    _Float16* __restrict__ og, int B, int N) {
  __shared__ _Float16 pT[H_][16][64];
  __shared__ _Float16 vT[H_][32][64];
  __shared__ float smask[48];

  int res = blockIdx.x;
  int b = res / N, n = res % N;
  int head = threadIdx.x >> 5;
  int lane = threadIdx.x & 31;
  int hi  = (lane & 16) ? 1 : 0;
  int lan = lane & 15;

  if (threadIdx.x < 48) {
    int c = threadIdx.x; float mv = 0.f;
    if (c < 3 * A_) {
      int s = c / A_, a = c - s * A_;
      int nn = n + s - 1;
      if (nn >= 0 && nn < N)
        mv = (amask[((size_t)b * N + nn) * A_ + a] != 0) ? 1.f : 0.f;
    }
    smask[c] = mv;
  }
  __syncthreads();

  size_t resbase = ((size_t)b * N + n) * A_ * D_;

  // Stage V^T (with per-slot f16 offset), zero-padded columns 42..63.
  for (int c = lane; c < 64; c += 32) {
    if (c < 3 * A_) {
      int s = c / A_, a = c - s * A_;
      int nn = min(max(n + s - 1, 0), N - 1);
      const _Float16* vrow = vg + (((size_t)b * N + nn) * A_ + a) * D_ + head * DH_;
      const _Float16* vo = voff + s * D_ + head * DH_;
#pragma unroll
      for (int d = 0; d < DH_; ++d)
        vT[head][d][c] = vrow[d] + vo[d];
    } else {
#pragma unroll
      for (int d = 0; d < DH_; ++d) vT[head][d][c] = (_Float16)0.f;
    }
  }

  // ---- scores = Q @ K^T (A rows >13 clamped: never stored, kept finite) ----
  int mq = lan > (A_ - 1) ? (A_ - 1) : lan;
  const _Float16* qrow = qg + resbase + (size_t)mq * D_ + head * DH_ + (hi ? 16 : 0);
  v16h afr;
#pragma unroll
  for (int p = 0; p < 8; ++p) {
    int k = 2 * p;
    afr[2 * p]     = qrow[k];
    afr[2 * p + 1] = qrow[k + 1];
  }

  v8f sc[3];
#pragma unroll
  for (int t = 0; t < 3; ++t)
#pragma unroll
    for (int i = 0; i < 8; ++i) sc[t][i] = 0.f;

#pragma unroll
  for (int t = 0; t < 3; ++t) {
    int c = lan + 16 * t;
    int cc = c > (3 * A_ - 1) ? (3 * A_ - 1) : c;
    int s = cc / A_, a = cc - s * A_;
    int nn = min(max(n + s - 1, 0), N - 1);
    const _Float16* krow = kg + (((size_t)b * N + nn) * A_ + a) * D_ + head * DH_ + hi * 16;
    const _Float16* ko = koff + s * D_ + head * DH_ + hi * 16;
    v16h bfr;
#pragma unroll
    for (int e = 0; e < 16; ++e) bfr[e] = krow[e] + ko[e];
    sc[t] = __builtin_amdgcn_wmma_f32_16x16x32_f16(false, afr, false, bfr,
                                                   (short)0, sc[t], false, false);
  }

  float m0 = smask[lan], m1 = smask[lan + 16], m2 = smask[lan + 32];
  const float scale = 0.17677669529663687f;  // 1/sqrt(32)

#pragma unroll
  for (int i = 0; i < 8; ++i) {
    float v0 = (m0 > 0.f) ? sc[0][i] * scale : NEG_BIG;
    float v1 = (m1 > 0.f) ? sc[1][i] * scale : NEG_BIG;
    float v2 = (m2 > 0.f) ? sc[2][i] * scale : NEG_BIG;
    float rm = fmaxf(v0, fmaxf(v1, v2));
#pragma unroll
    for (int msk = 1; msk < 16; msk <<= 1) rm = fmaxf(rm, __shfl_xor(rm, msk, 32));
    float e0 = (m0 > 0.f) ? __expf(v0 - rm) : 0.f;
    float e1 = (m1 > 0.f) ? __expf(v1 - rm) : 0.f;
    float e2 = (m2 > 0.f) ? __expf(v2 - rm) : 0.f;
    float rs = e0 + e1 + e2;
#pragma unroll
    for (int msk = 1; msk < 16; msk <<= 1) rs += __shfl_xor(rs, msk, 32);
    float inv = (rs > 0.f) ? (1.0f / rs) : 0.f;  // all-masked row -> p = 0
    int row = i + hi * 8;
    pT[head][row][lan]      = (_Float16)(e0 * inv);
    pT[head][row][lan + 16] = (_Float16)(e1 * inv);
    pT[head][row][lan + 32] = (_Float16)(e2 * inv);
    pT[head][row][lan + 48] = (_Float16)0.f;     // zero pad K=48..63
  }

  // ---- O = P @ V ----
  v8f oa[2];
#pragma unroll
  for (int j = 0; j < 2; ++j)
#pragma unroll
    for (int i = 0; i < 8; ++i) oa[j][i] = 0.f;

#pragma unroll
  for (int kc = 0; kc < 2; ++kc) {
    v16h pa;
#pragma unroll
    for (int p = 0; p < 8; ++p) {
      int k = kc * 32 + 2 * p + hi * 16;
      pa[2 * p]     = pT[head][lan][k];
      pa[2 * p + 1] = pT[head][lan][k + 1];
    }
#pragma unroll
    for (int nt = 0; nt < 2; ++nt) {
      int dh = lan + 16 * nt;
      int kb = kc * 32 + hi * 16;
      v16h vb;
#pragma unroll
      for (int e = 0; e < 16; ++e) vb[e] = vT[head][dh][kb + e];
      oa[nt] = __builtin_amdgcn_wmma_f32_16x16x32_f16(false, pa, false, vb,
                                                      (short)0, oa[nt], false, false);
    }
  }

#pragma unroll
  for (int nt = 0; nt < 2; ++nt) {
    int dh = lan + 16 * nt;
#pragma unroll
    for (int i = 0; i < 8; ++i) {
      int m = i + hi * 8;
      if (m < A_)
        og[resbase + (size_t)m * D_ + head * DH_ + dh] = (_Float16)oa[nt][i];
    }
  }
}

// ---------------------------------------------------------------------------
extern "C" void kernel_launch(void* const* d_in, const int* in_sizes, int n_in,
                              void* d_out, int out_size, void* d_ws, size_t ws_size,
                              hipStream_t stream) {
  (void)in_sizes; (void)n_in; (void)out_size; (void)ws_size;
  const float* x    = (const float*)d_in[0];
  const int*   amsk = (const int*)d_in[1];
  const float* n1w  = (const float*)d_in[2];
  const float* n1b  = (const float*)d_in[3];
  const float* qw   = (const float*)d_in[4];
  const float* kw   = (const float*)d_in[5];
  const float* vw   = (const float*)d_in[6];
  const float* ow   = (const float*)d_in[7];
  const float* offe = (const float*)d_in[8];
  const float* n2w  = (const float*)d_in[9];
  const float* n2b  = (const float*)d_in[10];
  const float* w1   = (const float*)d_in[11];
  const float* b1   = (const float*)d_in[12];
  const float* w2   = (const float*)d_in[13];
  const float* b2   = (const float*)d_in[14];

  const int B = 8, N = 512;
  const size_t M = (size_t)B * N * A_;              // 57344 atom rows
  const size_t hbytes = M * D_ * sizeof(_Float16);  // 28 MiB

  char* ws = (char*)d_ws;
  _Float16* h  = (_Float16*)(ws);                   // LN1 / LN2 activations
  _Float16* q  = (_Float16*)(ws + 1 * hbytes);
  _Float16* k  = (_Float16*)(ws + 2 * hbytes);
  _Float16* v  = (_Float16*)(ws + 3 * hbytes);
  _Float16* o  = (_Float16*)(ws + 4 * hbytes);
  _Float16* f  = q;                                 // FFN hidden reuses q..o (FF==4D)
  float*    x1 = (float*)(ws + 5 * hbytes);
  char* wp = ws + 5 * hbytes + M * D_ * sizeof(float);
  _Float16* qwf = (_Float16*)wp; wp += (size_t)D_ * D_ * 2;
  _Float16* kwf = (_Float16*)wp; wp += (size_t)D_ * D_ * 2;
  _Float16* vwf = (_Float16*)wp; wp += (size_t)D_ * D_ * 2;
  _Float16* owf = (_Float16*)wp; wp += (size_t)D_ * D_ * 2;
  _Float16* w1f = (_Float16*)wp; wp += (size_t)FF_ * D_ * 2;
  _Float16* w2f = (_Float16*)wp; wp += (size_t)FF_ * D_ * 2;
  _Float16* koff = (_Float16*)wp; wp += 3 * D_ * 2;
  _Float16* voff = (_Float16*)wp;

  // 1) cast weights to f16 + tiny offset projections
  cvt_f16_kernel<<<dim3(D_ * D_ / 256), dim3(256), 0, stream>>>(qw, qwf, D_ * D_);
  cvt_f16_kernel<<<dim3(D_ * D_ / 256), dim3(256), 0, stream>>>(kw, kwf, D_ * D_);
  cvt_f16_kernel<<<dim3(D_ * D_ / 256), dim3(256), 0, stream>>>(vw, vwf, D_ * D_);
  cvt_f16_kernel<<<dim3(D_ * D_ / 256), dim3(256), 0, stream>>>(ow, owf, D_ * D_);
  cvt_f16_kernel<<<dim3(FF_ * D_ / 256), dim3(256), 0, stream>>>(w1, w1f, FF_ * D_);
  cvt_f16_kernel<<<dim3(FF_ * D_ / 256), dim3(256), 0, stream>>>(w2, w2f, FF_ * D_);
  offproj_kernel<<<dim3(3), dim3(D_), 0, stream>>>(offe, kw, vw, koff, voff);

  dim3 blk(256);
  dim3 gLN((unsigned)(M / 8));
  dim3 g256((unsigned)(M / 256), D_ / 64);
  dim3 g1024((unsigned)(M / 256), FF_ / 64);

  // 2) LN1 -> h
  ln_cast_kernel<<<gLN, blk, 0, stream>>>(x, n1w, n1b, h, (int)M);
  // 3) Q/K/V flat GEMMs (K/V computed once per atom; offsets added in attn)
  gemm_wmma_kernel<0><<<g256, blk, 0, stream>>>(h, qwf, nullptr, nullptr, q, nullptr, (int)M, D_, D_);
  gemm_wmma_kernel<0><<<g256, blk, 0, stream>>>(h, kwf, nullptr, nullptr, k, nullptr, (int)M, D_, D_);
  gemm_wmma_kernel<0><<<g256, blk, 0, stream>>>(h, vwf, nullptr, nullptr, v, nullptr, (int)M, D_, D_);
  // 4) sliding-window attention
  attn_kernel<<<dim3(B * N), blk, 0, stream>>>(q, k, v, koff, voff, amsk, o, B, N);
  // 5) out-proj + residual -> x1 (fp32)
  gemm_wmma_kernel<2><<<g256, blk, 0, stream>>>(o, owf, nullptr, x, nullptr, x1, (int)M, D_, D_);
  // 6) LN2 -> h (reused)
  ln_cast_kernel<<<gLN, blk, 0, stream>>>(x1, n2w, n2b, h, (int)M);
  // 7) FFN1: bias + exact GELU -> f
  gemm_wmma_kernel<1><<<g1024, blk, 0, stream>>>(h, w1f, b1, nullptr, f, nullptr, (int)M, FF_, D_);
  // 8) FFN2: bias + residual(x1) -> d_out (fp32)
  gemm_wmma_kernel<3><<<g256, blk, 0, stream>>>(f, w2f, b2, x1, nullptr, (float*)d_out, (int)M, D_, FF_);
}
